// MMSA_37984690766445
// MI455X (gfx1250) — matrix operand order repaired and checked by reading, not compiled, not run on hardware
//
#include <hip/hip_runtime.h>
#include <hip/hip_bf16.h>

#define S   1024
#define DIM 1024
#define NH  16
#define HD  64
#define EH  64

typedef __attribute__((ext_vector_type(16))) __bf16 v16bf;
typedef __attribute__((ext_vector_type(8)))  float  v8f;

// ---------------- WMMA fragment loaders (16x32 bf16 tiles) ----------------
// A-matrix: lane L -> row M=L%16; 16 contiguous K elems kb..kb+15, kb=16*(L/16)
__device__ __forceinline__ v16bf load_fragA_bf16(const __bf16* tile, int ld) {
  const int lane = threadIdx.x & 31;
  const __bf16* p = tile + (lane & 15) * ld;
  const int kb = (lane >> 4) << 4;
  union { uint4 u; __bf16 h[8]; } lo, hi;
  lo.u = *reinterpret_cast<const uint4*>(p + kb);
  hi.u = *reinterpret_cast<const uint4*>(p + 8 + kb);
  v16bf f;
#pragma unroll
  for (int j = 0; j < 8; ++j) { f[j] = lo.h[j]; f[8 + j] = hi.h[j]; }
  return f;
}

__device__ __forceinline__ v16bf load_fragA_f32(const float* tile, int ld) {
  const int lane = threadIdx.x & 31;
  const float* p = tile + (lane & 15) * ld;
  const int kb = (lane >> 4) << 4;
  v16bf f;
#pragma unroll
  for (int j = 0; j < 8; ++j) {
    f[j]     = (__bf16)p[kb + j];
    f[8 + j] = (__bf16)p[8 + kb + j];
  }
  return f;
}

// B-matrix (we hold B^T row-major): lane L -> col N=L%16; 16 contiguous K at 16*(L/16)
__device__ __forceinline__ v16bf load_fragB_bf16(const __bf16* tileT, int ld) {
  const int lane = threadIdx.x & 31;
  const __bf16* p = tileT + (lane & 15) * ld + ((lane >> 4) << 4);
  union { uint4 u[2]; __bf16 h[16]; } d;
  d.u[0] = *reinterpret_cast<const uint4*>(p);
  d.u[1] = *reinterpret_cast<const uint4*>(p + 8);
  v16bf f;
#pragma unroll
  for (int j = 0; j < 16; ++j) f[j] = d.h[j];
  return f;
}

__device__ __forceinline__ v16bf load_fragB_f32(const float* tileT, int ld) {
  const int lane = threadIdx.x & 31;
  const float* p = tileT + (lane & 15) * ld + ((lane >> 4) << 4);
  v16bf f;
#pragma unroll
  for (int j = 0; j < 16; ++j) f[j] = (__bf16)p[j];
  return f;
}

__device__ __forceinline__ v8f wmma_bf16(v16bf a, v16bf b, v8f c) {
  return __builtin_amdgcn_wmma_f32_16x16x32_bf16(false, a, false, b, (short)0, c,
                                                 false, false);
}

// ---------------- Kernel A: QKV projection (bf16 WMMA, fp32 in) ----------------
// Each wave computes a 16x64 strip (4 accumulators): A fragment loaded once per
// K-step, 4 independent WMMAs issued back-to-back.
__global__ void __launch_bounds__(128)
k_qkv(const float* __restrict__ x, const float* __restrict__ w,
      const float* __restrict__ bias,
      __bf16* __restrict__ qbf, __bf16* __restrict__ kbf, __bf16* __restrict__ vt) {
  const int wid = threadIdx.x >> 5, lane = threadIdx.x & 31;
  const int n0 = (blockIdx.x * 4 + wid) * 64;   // 64-col strip per wave
  const int m0 = blockIdx.y * 16;
  const float* arow = x + (size_t)m0 * DIM;
  v8f acc[4] = {{}, {}, {}, {}};
#pragma unroll 1
  for (int kk = 0; kk < DIM; kk += 32) {
    const v16bf a = load_fragA_f32(arow + kk, DIM);
#pragma unroll
    for (int t = 0; t < 4; ++t)
      acc[t] = wmma_bf16(a, load_fragB_f32(w + (size_t)(n0 + t * 16) * DIM + kk, DIM),
                         acc[t]);
  }
#pragma unroll
  for (int t = 0; t < 4; ++t) {
    const int col = n0 + t * 16 + (lane & 15);
    const float bv = bias[col];
    const int which = col >> 10;            // 0=q 1=k 2=v
    const int h = (col >> 6) & (NH - 1);
    const int d = col & (HD - 1);
#pragma unroll
    for (int i = 0; i < 8; ++i) {
      const int m = m0 + i + ((lane >> 4) << 3);
      const float val = acc[t][i] + bv;
      if (which == 0)      qbf[((size_t)h * S + m) * HD + d] = (__bf16)(val * 0.125f);
      else if (which == 1) kbf[((size_t)h * S + m) * HD + d] = (__bf16)val;
      else                 vt[((size_t)h * HD + d) * S + m]  = (__bf16)val;
    }
  }
}

// ---------------- Kernel B: per-head raw scores S[h][q][k] = (Q*0.125) K^T ----
// Block = 4 waves covering a 64q x 64k macro-tile; the shared 64x64 bf16 K-strip
// is staged to LDS with the CDNA5 async-copy path, then fragments come from LDS.
__global__ void __launch_bounds__(128)
k_scores(const __bf16* __restrict__ qbf, const __bf16* __restrict__ kbf,
         __bf16* __restrict__ sraw) {
  __shared__ alignas(16) __bf16 ldsk[64 * HD];          // 8 KB K-strip
  const int wid = threadIdx.x >> 5, lane = threadIdx.x & 31;
  const int tid = threadIdx.x;
  const int h = blockIdx.z;
  const int kblk = blockIdx.x;                           // 64-key strip
  const int qblk = blockIdx.y;                           // 64-query strip

  // async-copy 8 KB K-strip to LDS: 128 lanes x 4 x b128
  {
    const __bf16* gsrc = kbf + ((size_t)h * S + kblk * 64) * HD;
    const unsigned ldsbase = (unsigned)(size_t)(void*)ldsk;    // LDS = addr[31:0]
    const unsigned long long gbase = (unsigned long long)(size_t)gsrc;
#pragma unroll
    for (int i = 0; i < 4; ++i) {
      const unsigned off = (unsigned)tid * 16u + (unsigned)i * 2048u;
      asm volatile("global_load_async_to_lds_b128 %0, %1, %2 offset:0"
                   :: "v"(ldsbase + off), "v"(off), "s"(gbase)
                   : "memory");
    }
    asm volatile("s_wait_asynccnt 0" ::: "memory");
  }
  __syncthreads();

  const __bf16* qh = qbf + ((size_t)h * S + qblk * 64 + wid * 16) * HD;
  v8f acc[4] = {{}, {}, {}, {}};
#pragma unroll
  for (int kk = 0; kk < HD; kk += 32) {
    const v16bf a = load_fragA_bf16(qh + kk, HD);
#pragma unroll
    for (int t = 0; t < 4; ++t)
      acc[t] = wmma_bf16(a, load_fragB_bf16(ldsk + t * 16 * HD + kk, HD), acc[t]);
  }
#pragma unroll
  for (int t = 0; t < 4; ++t) {
    __bf16* out = sraw + ((size_t)h << 20) +
                  (size_t)(kblk * 64 + t * 16 + (lane & 15));
#pragma unroll
    for (int i = 0; i < 8; ++i) {
      const int m = qblk * 64 + wid * 16 + i + ((lane >> 4) << 3);
      out[(size_t)m * S] = (__bf16)acc[t][i];
    }
  }
}

// ---------------- Kernel C: fused head-mix -> softmax -> head-mix-back --------
// One block per q row; 16 raw head rows live in LDS (64KB); expanded 64-head
// tensor never exists in memory.
__global__ void __launch_bounds__(256)
k_mix_softmax(const __bf16* __restrict__ sraw,
              const float* __restrict__ ap1w, const float* __restrict__ ap1b,
              const float* __restrict__ ap2w, const float* __restrict__ ap2b,
              __bf16* __restrict__ post) {
  extern __shared__ float raw[];          // NH * S floats = 64 KB
  __shared__ float red[8];
  const int q = blockIdx.x;
  const int tid = threadIdx.x;

  for (int idx = tid; idx < NH * S; idx += 256) {
    const int h = idx >> 10, kcol = idx & (S - 1);
    raw[idx] = (float)sraw[((size_t)h << 20) + (size_t)q * S + kcol];
  }
  __syncthreads();

  float outacc[NH][4];
#pragma unroll
  for (int h = 0; h < NH; ++h)
#pragma unroll
    for (int j = 0; j < 4; ++j) outacc[h][j] = 0.f;

  for (int e = 0; e < EH; ++e) {
    float v[4];
#pragma unroll
    for (int j = 0; j < 4; ++j) {
      const int kcol = tid + 256 * j;
      float s = ap1b[e];
#pragma unroll
      for (int h = 0; h < NH; ++h) s = fmaf(ap1w[e * NH + h], raw[h * S + kcol], s);
      v[j] = s;
    }
    // block-wide max over the 1024 key columns
    float m = fmaxf(fmaxf(v[0], v[1]), fmaxf(v[2], v[3]));
#pragma unroll
    for (int o = 16; o > 0; o >>= 1) m = fmaxf(m, __shfl_xor(m, o, 32));
    if ((tid & 31) == 0) red[tid >> 5] = m;
    __syncthreads();
    m = red[0];
#pragma unroll
    for (int i = 1; i < 8; ++i) m = fmaxf(m, red[i]);
    __syncthreads();
    // block-wide sum of exp
    float p[4], ssum = 0.f;
#pragma unroll
    for (int j = 0; j < 4; ++j) { p[j] = __expf(v[j] - m); ssum += p[j]; }
#pragma unroll
    for (int o = 16; o > 0; o >>= 1) ssum += __shfl_xor(ssum, o, 32);
    if ((tid & 31) == 0) red[tid >> 5] = ssum;
    __syncthreads();
    ssum = 0.f;
#pragma unroll
    for (int i = 0; i < 8; ++i) ssum += red[i];
    __syncthreads();
    const float inv = 1.0f / ssum;
    // contract expanded head e back into 16 output heads
#pragma unroll
    for (int j = 0; j < 4; ++j) {
      const float pe = p[j] * inv;
#pragma unroll
      for (int h = 0; h < NH; ++h)
        outacc[h][j] = fmaf(ap2w[h * EH + e], pe, outacc[h][j]);
    }
  }
#pragma unroll
  for (int h = 0; h < NH; ++h) {
    const float b2 = ap2b[h];
#pragma unroll
    for (int j = 0; j < 4; ++j) {
      const int kcol = tid + 256 * j;
      post[((size_t)h << 20) + (size_t)q * S + kcol] = (__bf16)(outacc[h][j] + b2);
    }
  }
}

// ---------------- Kernel D: out_h = P[h] @ V[h]  (V stored transposed) -------
// Wave computes 16q x 64d (all of N) with 4 accumulators.
__global__ void __launch_bounds__(128)
k_attnv(const __bf16* __restrict__ post, const __bf16* __restrict__ vt,
        __bf16* __restrict__ merged) {
  const int wid = threadIdx.x >> 5, lane = threadIdx.x & 31;
  const int h = blockIdx.z;
  const int qt0 = blockIdx.y * 64 + wid * 16;
  const __bf16* arow = post + ((size_t)h << 20) + (size_t)qt0 * S;
  const __bf16* vh = vt + (size_t)h * HD * S;
  v8f acc[4] = {{}, {}, {}, {}};
#pragma unroll 1
  for (int kk = 0; kk < S; kk += 32) {
    const v16bf a = load_fragA_bf16(arow + kk, S);
#pragma unroll
    for (int t = 0; t < 4; ++t)
      acc[t] = wmma_bf16(a, load_fragB_bf16(vh + (size_t)(t * 16) * S + kk, S), acc[t]);
  }
#pragma unroll
  for (int t = 0; t < 4; ++t) {
    const int dcol = h * HD + t * 16 + (lane & 15);
#pragma unroll
    for (int i = 0; i < 8; ++i) {
      const int m = qt0 + i + ((lane >> 4) << 3);
      merged[(size_t)m * DIM + dcol] = (__bf16)acc[t][i];
    }
  }
}

// ---------------- Kernel E: output projection (bf16 A x fp32 W) -------------
__global__ void __launch_bounds__(128)
k_proj(const __bf16* __restrict__ merged, const float* __restrict__ w,
       const float* __restrict__ bias, float* __restrict__ out) {
  const int wid = threadIdx.x >> 5, lane = threadIdx.x & 31;
  const int n0 = (blockIdx.x * 4 + wid) * 64, m0 = blockIdx.y * 16;
  const __bf16* arow = merged + (size_t)m0 * DIM;
  v8f acc[4] = {{}, {}, {}, {}};
#pragma unroll 1
  for (int kk = 0; kk < DIM; kk += 32) {
    const v16bf a = load_fragA_bf16(arow + kk, DIM);
#pragma unroll
    for (int t = 0; t < 4; ++t)
      acc[t] = wmma_bf16(a, load_fragB_f32(w + (size_t)(n0 + t * 16) * DIM + kk, DIM),
                         acc[t]);
  }
#pragma unroll
  for (int t = 0; t < 4; ++t) {
    const int col = n0 + t * 16 + (lane & 15);
    const float bv = bias[col];
#pragma unroll
    for (int i = 0; i < 8; ++i) {
      const int m = m0 + i + ((lane >> 4) << 3);
      out[(size_t)m * DIM + col] = acc[t][i] + bv;
    }
  }
}

extern "C" void kernel_launch(void* const* d_in, const int* in_sizes, int n_in,
                              void* d_out, int out_size, void* d_ws, size_t ws_size,
                              hipStream_t stream) {
  (void)in_sizes; (void)n_in; (void)out_size; (void)ws_size;
  const float* x      = (const float*)d_in[0];
  const float* qkv_w  = (const float*)d_in[1];
  const float* qkv_b  = (const float*)d_in[2];
  const float* proj_w = (const float*)d_in[3];
  const float* proj_b = (const float*)d_in[4];
  const float* ap1_w  = (const float*)d_in[5];
  const float* ap1_b  = (const float*)d_in[6];
  const float* ap2_w  = (const float*)d_in[7];
  const float* ap2_b  = (const float*)d_in[8];
  float* out = (float*)d_out;

  // bf16 workspace layout (element offsets, 1 Mi elems = 2 MB):
  //  [0,1M)   Q scaled     [1M,2M) K        [2M,3M) V^T
  //  [3M,19M) raw scores   [19M,35M) post-mix probs   [35M,36M) merged heads
  __bf16* wsb = (__bf16*)d_ws;
  const size_t M1 = (size_t)1 << 20;
  __bf16* qbf    = wsb;
  __bf16* kbf    = wsb + 1 * M1;
  __bf16* vt     = wsb + 2 * M1;
  __bf16* sraw   = wsb + 3 * M1;
  __bf16* post   = wsb + 19 * M1;
  __bf16* merged = wsb + 35 * M1;

  k_qkv        <<<dim3(12, 64),     128, 0, stream>>>(x, qkv_w, qkv_b, qbf, kbf, vt);
  k_scores     <<<dim3(16, 16, 16), 128, 0, stream>>>(qbf, kbf, sraw);
  k_mix_softmax<<<dim3(1024), 256, NH * S * sizeof(float), stream>>>(
      sraw, ap1_w, ap1_b, ap2_w, ap2_b, post);
  k_attnv      <<<dim3(1, 16, 16),  128, 0, stream>>>(post, vt, merged);
  k_proj       <<<dim3(4, 64),      128, 0, stream>>>(merged, proj_w, proj_b, out);
}
